// QNetwork_40149354283753
// MI455X (gfx1250) — matrix operand order repaired and checked by reading, not compiled, not run on hardware
//
#include <hip/hip_runtime.h>
#include <hip/hip_bf16.h>

typedef _Float16 half_t;
typedef __attribute__((ext_vector_type(16))) _Float16 v16h;
typedef __attribute__((ext_vector_type(8)))  _Float16 v8h;
typedef __attribute__((ext_vector_type(8)))  float    v8f;

#define BATCH 16
#define SEQ   32
#define ROWS  512      // BATCH*SEQ
#define HWDIM 84
#define PIX   7056     // 84*84
#define HP    86       // padded (halo) dim
#define PPIX  7396     // 86*86
#define C0    32
#define C1    64
#define HIDD  256
#define NHEAD 4
#define HDIM  64
#define EPSF  1e-6f
#define MTILES 441     // 7056/16
#define NTILE_TOT (ROWS*MTILES)   // 225792 (even)
#define NBORD 340      // border pixels per 86x86 image: 2*86 + 2*84

// ---------------------------------------------------------------------------
// Weight prep. Weights pre-shuffled into WMMA *fragment order* so the conv
// kernels fetch each 16-half B fragment as two b128 loads:
//   fragment K-mapping (16-bit B, 32x16): K = (e<8 ? e : 8+e) + 8*(lane>=16), n = lane%16.
// wf0: conv0 [ks][nt][lane][e]  (K padded 36->64)
// wf1: conv1 [tap][nt][lane][e]
// Also zeroes the pooled accumulator and the h0 halo border.
// ---------------------------------------------------------------------------
__global__ void prep_kernel(const float* __restrict__ w0, const float* __restrict__ w1,
                            half_t* __restrict__ wf0, half_t* __restrict__ wf1,
                            float* __restrict__ pooled, half_t* __restrict__ h0) {
    int i = blockIdx.x * blockDim.x + threadIdx.x;
    if (i < 2 * 2 * 32 * 16) {                       // conv0 fragments
        int e = i & 15, lane = (i >> 4) & 31, nt = (i >> 9) & 1, ks = i >> 10;
        int K = (e < 8 ? e : 8 + e) + ((lane >= 16) ? 8 : 0) + ks * 32;
        int n = nt * 16 + (lane & 15);
        wf0[i] = (K < 36) ? (half_t)w0[K * 32 + n] : (half_t)0.f;
    }
    if (i < 9 * 4 * 32 * 16) {                       // conv1 fragments
        int e = i & 15, lane = (i >> 4) & 31, nt = (i >> 9) & 3, tap = i >> 11;
        int cin = (e < 8 ? e : 8 + e) + ((lane >= 16) ? 8 : 0);
        int n = nt * 16 + (lane & 15);
        wf1[i] = (half_t)w1[(tap * 32 + cin) * 64 + n];
    }
    if (i < ROWS * C1) pooled[i] = 0.f;
    if (i < ROWS * NBORD) {                          // zero h0 halo border
        int img = i / NBORD, bi = i % NBORD;
        int r, c;
        if (bi < 86)       { r = 0;            c = bi; }
        else if (bi < 172) { r = 85;           c = bi - 86; }
        else if (bi < 256) { r = bi - 172 + 1; c = 0; }
        else               { r = bi - 256 + 1; c = 85; }
        v8h z;
        #pragma unroll
        for (int e = 0; e < 8; ++e) z[e] = (half_t)0.f;
        v8h* dst = (v8h*)(h0 + ((long)img * PPIX + r * HP + c) * 32);
        dst[0] = z; dst[1] = z; dst[2] = z; dst[3] = z;
    }
}

__device__ __forceinline__ float4 load_tap4(const float* __restrict__ x,
                                            int img, int rr, int cc) {
    float4 v = make_float4(0.f, 0.f, 0.f, 0.f);
    if (rr >= 0 && rr < HWDIM && cc >= 0 && cc < HWDIM)
        v = *(const float4*)(x + (((img * HWDIM + rr) * HWDIM) + cc) * 4);
    return v;
}

// ---------------------------------------------------------------------------
// conv0: implicit GEMM, M=512*7056 pixels, N=32, K=36 (pad 64, 2 WMMA steps).
// A fragment = whole taps (4 contiguous fp32 ch -> float4 vector loads).
// Fused: +bias, LayerNorm over 32 channels, ReLU; f16 store into 86x86 halo.
// Pixel (row,col) derived from one div per wave + conditional wrap (no
// per-row magic-number division chains).
// ---------------------------------------------------------------------------
__global__ __launch_bounds__(256) void conv0_kernel(
    const float* __restrict__ x, const half_t* __restrict__ wf0,
    const float* __restrict__ bias, const float* __restrict__ lns,
    const float* __restrict__ lnb, half_t* __restrict__ h0)
{
    const int lane = threadIdx.x & 31;
    const int wv   = threadIdx.x >> 5;
    const int tile = blockIdx.x * 8 + wv;
    if (tile >= NTILE_TOT) return;
    const int img = tile / MTILES;
    const int tp  = (tile % MTILES) * 16;
    const int m   = lane & 15;
    const int hi  = lane >> 4;
    const int n0  = lane & 15;

    const int r0 = tp / HWDIM, c0 = tp % HWDIM;      // one div per wave
    int pc = c0 + m, pr = r0;
    if (pc >= HWDIM) { pc -= HWDIM; pr += 1; }       // 16 pixels wrap <= once

    // --- A fragments. Per-lane taps: {2hi, 2hi+1, 4+2hi, 5+2hi} (ks0), tap8 (ks1,hi=0)
    // tap t -> (dy,dx) = (t/3-1, t%3-1); resolved by hi-select (no runtime div).
    float4 f0 = load_tap4(x, img, pr - 1,            pc + (hi ?  1 : -1));   // tap 0 / 2
    float4 f1 = load_tap4(x, img, pr + (hi ? 0 : -1), pc + (hi ? -1 :  0));  // tap 1 / 3
    float4 f2 = load_tap4(x, img, pr + (hi ? 1 : 0),  pc + (hi ? -1 :  0));  // tap 4 / 6
    float4 f3 = load_tap4(x, img, pr + (hi ? 1 : 0),  pc + (hi ? 0 :  1));   // tap 5 / 7
    v16h a0;
    a0[0] = (half_t)f0.x; a0[1] = (half_t)f0.y; a0[2]  = (half_t)f0.z; a0[3]  = (half_t)f0.w;
    a0[4] = (half_t)f1.x; a0[5] = (half_t)f1.y; a0[6]  = (half_t)f1.z; a0[7]  = (half_t)f1.w;
    a0[8] = (half_t)f2.x; a0[9] = (half_t)f2.y; a0[10] = (half_t)f2.z; a0[11] = (half_t)f2.w;
    a0[12] = (half_t)f3.x; a0[13] = (half_t)f3.y; a0[14] = (half_t)f3.z; a0[15] = (half_t)f3.w;
    v16h a1;
    #pragma unroll
    for (int e = 0; e < 16; ++e) a1[e] = (half_t)0.f;
    if (hi == 0) {                // ks1: K=32..35 = tap 8 (dy=+1,dx=+1), low half only
        float4 f8 = load_tap4(x, img, pr + 1, pc + 1);
        a1[0] = (half_t)f8.x; a1[1] = (half_t)f8.y; a1[2] = (half_t)f8.z; a1[3] = (half_t)f8.w;
    }

    v8f acc[2];
    #pragma unroll
    for (int nt = 0; nt < 2; ++nt)
        #pragma unroll
        for (int r = 0; r < 8; ++r) acc[nt][r] = 0.f;

    #pragma unroll
    for (int ks = 0; ks < 2; ++ks) {
        v16h a = ks ? a1 : a0;
        #pragma unroll
        for (int nt = 0; nt < 2; ++nt) {
            const v8h* wp = (const v8h*)(wf0 + (((ks * 2 + nt) * 32 + lane) << 4));
            v8h b1 = wp[0], b2 = wp[1];
            v16h bf;
            #pragma unroll
            for (int e = 0; e < 8; ++e) { bf[e] = b1[e]; bf[8 + e] = b2[e]; }
            acc[nt] = __builtin_amdgcn_wmma_f32_16x16x32_f16(
                false, a, false, bf, (short)0, acc[nt], false, false);
        }
    }

    float bn[2], sn[2], bb[2];
    #pragma unroll
    for (int nt = 0; nt < 2; ++nt) {
        int n = nt * 16 + n0;
        bn[nt] = bias[n]; sn[nt] = lns[n]; bb[nt] = lnb[n];
    }
    #pragma unroll
    for (int r = 0; r < 8; ++r) {
        float v0 = acc[0][r] + bn[0];
        float v1 = acc[1][r] + bn[1];
        float s = v0 + v1, sq = v0 * v0 + v1 * v1;
        #pragma unroll
        for (int msk = 1; msk < 16; msk <<= 1) {      // reduce within 16-lane half
            s  += __shfl_xor(s,  msk, 32);
            sq += __shfl_xor(sq, msk, 32);
        }
        float mean = s * (1.f / 32.f);
        float var  = sq * (1.f / 32.f) - mean * mean;
        float inv  = rsqrtf(var + EPSF);
        float o0 = (v0 - mean) * inv * sn[0] + bb[0]; o0 = o0 > 0.f ? o0 : 0.f;
        float o1 = (v1 - mean) * inv * sn[1] + bb[1]; o1 = o1 > 0.f ? o1 : 0.f;
        int mm = r + hi * 8;
        int cS = c0 + mm, rS = r0;
        if (cS >= HWDIM) { cS -= HWDIM; rS += 1; }
        long base = ((long)img * PPIX + (rS + 1) * HP + (cS + 1)) * 32;
        h0[base + n0]      = (half_t)o0;
        h0[base + 16 + n0] = (half_t)o1;
    }
}

// ---------------------------------------------------------------------------
// conv1: implicit GEMM, M=512*7056, N=64, K=288 (9 taps x 32ch = 9 WMMA steps).
// Each wave computes TWO 16-pixel M-tiles so every B fragment loaded from LDS
// feeds two WMMAs (halves LDS traffic per matrix op). Halo layout keeps A
// loads branchless b128 pairs. Fused: +bias, LN(64), ReLU, mean-pool atomics.
// ---------------------------------------------------------------------------
__global__ __launch_bounds__(256) void conv1_kernel(
    const half_t* __restrict__ h0, const half_t* __restrict__ wf1,
    const float* __restrict__ bias, const float* __restrict__ lns,
    const float* __restrict__ lnb, float* __restrict__ pooled)
{
    __shared__ half_t ws[9 * 4 * 32 * 16];
    {
        const v8h* src = (const v8h*)wf1;
        v8h* dst = (v8h*)ws;
        for (int i = threadIdx.x; i < 9 * 4 * 32 * 16 / 8; i += 256) dst[i] = src[i];
    }
    __syncthreads();

    const int lane  = threadIdx.x & 31;
    const int wv    = threadIdx.x >> 5;
    const int wtile = (blockIdx.x * 8 + wv) * 2;     // first of 2 M-tiles
    if (wtile >= NTILE_TOT) return;
    const int m  = lane & 15;
    const int hi = lane >> 4;
    const int n0 = lane & 15;
    const int chunkoff = hi * 8;

    int  imgA[2];
    long baseA[2];
    #pragma unroll
    for (int st = 0; st < 2; ++st) {
        int tile = wtile + st;
        int img  = tile / MTILES;
        int tp   = (tile % MTILES) * 16;
        imgA[st] = img;
        int r0 = tp / HWDIM, c0 = tp % HWDIM;
        int pc = c0 + m, pr = r0;
        if (pc >= HWDIM) { pc -= HWDIM; pr += 1; }
        baseA[st] = ((long)img * PPIX + pr * HP + pc) * 32;  // halo origin: tap adds 0..2
    }

    v8f acc[2][4];
    #pragma unroll
    for (int st = 0; st < 2; ++st)
        #pragma unroll
        for (int nt = 0; nt < 4; ++nt)
            #pragma unroll
            for (int r = 0; r < 8; ++r) acc[st][nt][r] = 0.f;

    #pragma unroll
    for (int tap = 0; tap < 9; ++tap) {
        v16h a[2];
        #pragma unroll
        for (int st = 0; st < 2; ++st) {
            const half_t* src = h0 + baseA[st] + ((tap / 3) * HP + (tap % 3)) * 32;
            v8h ca = *(const v8h*)(src + chunkoff);        // K = hi*8 .. hi*8+7
            v8h cb = *(const v8h*)(src + 16 + chunkoff);   // K = 16+hi*8 ..
            #pragma unroll
            for (int e = 0; e < 8; ++e) { a[st][e] = ca[e]; a[st][8 + e] = cb[e]; }
        }
        #pragma unroll
        for (int nt = 0; nt < 4; ++nt) {
            const v8h* wp = (const v8h*)(ws + (((tap * 4 + nt) * 32 + lane) << 4));
            v8h b1 = wp[0], b2 = wp[1];
            v16h bf;
            #pragma unroll
            for (int e = 0; e < 8; ++e) { bf[e] = b1[e]; bf[8 + e] = b2[e]; }
            acc[0][nt] = __builtin_amdgcn_wmma_f32_16x16x32_f16(
                false, a[0], false, bf, (short)0, acc[0][nt], false, false);
            acc[1][nt] = __builtin_amdgcn_wmma_f32_16x16x32_f16(
                false, a[1], false, bf, (short)0, acc[1][nt], false, false);
        }
    }

    float bn[4], sn[4], bb[4];
    #pragma unroll
    for (int nt = 0; nt < 4; ++nt) {
        int n = nt * 16 + n0;
        bn[nt] = bias[n]; sn[nt] = lns[n]; bb[nt] = lnb[n];
    }
    #pragma unroll
    for (int st = 0; st < 2; ++st) {
        float pool[4] = {0.f, 0.f, 0.f, 0.f};
        #pragma unroll
        for (int r = 0; r < 8; ++r) {
            float v[4]; float s = 0.f, sq = 0.f;
            #pragma unroll
            for (int nt = 0; nt < 4; ++nt) {
                v[nt] = acc[st][nt][r] + bn[nt];
                s += v[nt]; sq += v[nt] * v[nt];
            }
            #pragma unroll
            for (int msk = 1; msk < 16; msk <<= 1) {
                s  += __shfl_xor(s,  msk, 32);
                sq += __shfl_xor(sq, msk, 32);
            }
            float mean = s * (1.f / 64.f);
            float var  = sq * (1.f / 64.f) - mean * mean;
            float inv  = rsqrtf(var + EPSF);
            #pragma unroll
            for (int nt = 0; nt < 4; ++nt) {
                float o = (v[nt] - mean) * inv * sn[nt] + bb[nt];
                o = o > 0.f ? o : 0.f;
                pool[nt] += o;
            }
        }
        #pragma unroll
        for (int nt = 0; nt < 4; ++nt) {
            float t = pool[nt] + __shfl_xor(pool[nt], 16, 32); // both row-halves (same n)
            if (lane < 16) atomicAdd(&pooled[imgA[st] * C1 + nt * 16 + n0], t);
        }
    }
}

// ---------------------------------------------------------------------------
// Generic fp32 GEMM: C[m,n] = relu?( aScale*sum_k A[m,k]B[k,n] + bias[n] ) * pad[m]
// (tail of the network: ~1 GFLOP total)
// ---------------------------------------------------------------------------
__global__ void gemm_f32(const float* __restrict__ A, const float* __restrict__ Bm,
                         const float* __restrict__ bias, const unsigned char* __restrict__ rowmask,
                         float* __restrict__ C, int M, int N, int K, float aScale, int relu)
{
    int idx = blockIdx.x * blockDim.x + threadIdx.x;
    if (idx >= M * N) return;
    int mr = idx / N, n = idx % N;
    const float* a = A + (long)mr * K;
    float acc = 0.f;
    for (int k = 0; k < K; ++k) acc += a[k] * Bm[(long)k * N + n];
    acc *= aScale;
    if (bias) acc += bias[n];
    if (relu) acc = acc > 0.f ? acc : 0.f;
    if (rowmask) acc *= rowmask[mr] ? 1.f : 0.f;
    C[idx] = acc;
}

// ---------------------------------------------------------------------------
// LayerNorm over 256 features, one block per row, optional residual add
// (res may alias out; each thread reads-then-writes only its own element).
// ---------------------------------------------------------------------------
__global__ __launch_bounds__(256) void ln256_kernel(
    const float* __restrict__ x, const float* __restrict__ s, const float* __restrict__ b,
    const float* __restrict__ res, float* __restrict__ out)
{
    int row = blockIdx.x, t = threadIdx.x;
    float v = x[row * HIDD + t];
    float ss = v, sq = v * v;
    #pragma unroll
    for (int msk = 1; msk < 32; msk <<= 1) {
        ss += __shfl_xor(ss, msk, 32);
        sq += __shfl_xor(sq, msk, 32);
    }
    __shared__ float rs[8], rq[8];
    __shared__ float smean, sinv;
    int lane = t & 31, wid = t >> 5;
    if (lane == 0) { rs[wid] = ss; rq[wid] = sq; }
    __syncthreads();
    if (t == 0) {
        float S = 0.f, Q = 0.f;
        for (int i = 0; i < 8; ++i) { S += rs[i]; Q += rq[i]; }
        float mean = S * (1.f / 256.f);
        float var  = Q * (1.f / 256.f) - mean * mean;
        smean = mean; sinv = rsqrtf(var + EPSF);
    }
    __syncthreads();
    float o = (v - smean) * sinv * s[t] + b[t];
    if (res) o += res[row * HIDD + t];
    out[row * HIDD + t] = o;
}

// ---------------------------------------------------------------------------
// Attention: one block (64 threads) per (batch, head, qpos). Wave 0 computes
// RoPE'd logits (lane = key pos) + wave-wide softmax; both waves form output.
// qkv row layout: [q(256) | k(256) | v(256)].
// ---------------------------------------------------------------------------
__global__ __launch_bounds__(64) void attn_kernel(
    const float* __restrict__ qkv, const unsigned char* __restrict__ pad,
    float* __restrict__ y)
{
    int bid  = blockIdx.x;
    int b    = bid / (NHEAD * SEQ);
    int hh   = (bid / SEQ) % NHEAD;
    int qpos = bid % SEQ;
    int t    = threadIdx.x;
    __shared__ float attnS[SEQ];

    if (t < SEQ) {
        int kpos = t;
        const float* qv = qkv + (long)(b * SEQ + qpos) * 768 + hh * HDIM;
        const float* kv = qkv + (long)(b * SEQ + kpos) * 768 + 256 + hh * HDIM;
        float dot = 0.f;
        #pragma unroll
        for (int j = 0; j < 32; ++j) {
            // inv_freq = 10000^(-2j/64) = exp(-j * ln(1e4)/32)
            float invf = expf(-0.28782313662425572f * (float)j);
            float aq = (float)qpos * invf, ak = (float)kpos * invf;
            float cq = cosf(aq), sqn = sinf(aq);
            float ck = cosf(ak), skn = sinf(ak);
            float qe = qv[2 * j], qo = qv[2 * j + 1];
            float ke = kv[2 * j], ko = kv[2 * j + 1];
            float qre = qe * cq - qo * sqn, qro = qo * cq + qe * sqn;
            float kre = ke * ck - ko * skn, kro = ko * ck + ke * skn;
            dot += qre * kre + qro * kro;
        }
        float logit = dot * 0.125f;                       // HD^-0.5
        bool ok = (kpos <= qpos) && (pad[b * SEQ + kpos] != 0);
        logit = ok ? logit : -1e9f;
        float mx = logit;
        #pragma unroll
        for (int msk = 1; msk < 32; msk <<= 1) mx = fmaxf(mx, __shfl_xor(mx, msk, 32));
        float e = expf(logit - mx);
        float sum = e;
        #pragma unroll
        for (int msk = 1; msk < 32; msk <<= 1) sum += __shfl_xor(sum, msk, 32);
        attnS[kpos] = e / sum;
    }
    __syncthreads();
    float accv = 0.f;
    for (int k = 0; k < SEQ; ++k)
        accv += attnS[k] * qkv[(long)(b * SEQ + k) * 768 + 512 + hh * HDIM + t];
    y[(long)(b * SEQ + qpos) * HIDD + hh * HDIM + t] = accv;
}

// ---------------------------------------------------------------------------
extern "C" void kernel_launch(void* const* d_in, const int* in_sizes, int n_in,
                              void* d_out, int out_size, void* d_ws, size_t ws_size,
                              hipStream_t stream)
{
    const float* x        = (const float*)d_in[0];
    const unsigned char* pad = (const unsigned char*)d_in[1];   // jnp.bool_ -> 1 byte
    const float* conv0_w  = (const float*)d_in[2];
    const float* conv0_b  = (const float*)d_in[3];
    const float* ln_c0_s  = (const float*)d_in[4];
    const float* ln_c0_b  = (const float*)d_in[5];
    const float* conv1_w  = (const float*)d_in[6];
    const float* conv1_b  = (const float*)d_in[7];
    const float* ln_c1_s  = (const float*)d_in[8];
    const float* ln_c1_b  = (const float*)d_in[9];
    const float* proj_w   = (const float*)d_in[10];
    const float* proj_b   = (const float*)d_in[11];
    const float* ln_p_s   = (const float*)d_in[12];
    const float* ln_p_b   = (const float*)d_in[13];
    const float* qkv_w    = (const float*)d_in[14];
    const float* qkv_b    = (const float*)d_in[15];
    const float* out_w    = (const float*)d_in[16];
    const float* out_b    = (const float*)d_in[17];
    const float* ln_a_s   = (const float*)d_in[18];
    const float* ln_a_b   = (const float*)d_in[19];
    const float* up_w     = (const float*)d_in[20];
    const float* down_w   = (const float*)d_in[21];
    const float* ln_m_s   = (const float*)d_in[22];
    const float* ln_m_b   = (const float*)d_in[23];
    const float* head_w   = (const float*)d_in[24];
    const float* head_b   = (const float*)d_in[25];

    char* wsb = (char*)d_ws;
    size_t off = 0;
    auto alloc = [&](size_t bytes) -> void* {
        void* p = wsb + off;
        off = (off + bytes + 255) & ~(size_t)255;
        return p;
    };
    half_t* h0     = (half_t*)alloc((size_t)ROWS * PPIX * C0 * sizeof(half_t)); // 242 MB (halo)
    half_t* wf0    = (half_t*)alloc(2 * 2 * 32 * 16 * sizeof(half_t));
    half_t* wf1    = (half_t*)alloc(9 * 4 * 32 * 16 * sizeof(half_t));
    float* pooled  = (float*)alloc((size_t)ROWS * C1 * sizeof(float));
    float* hpre    = (float*)alloc((size_t)ROWS * HIDD * sizeof(float));
    float* h       = (float*)alloc((size_t)ROWS * HIDD * sizeof(float));
    float* qkv     = (float*)alloc((size_t)ROWS * 3 * HIDD * sizeof(float));
    float* ybuf    = (float*)alloc((size_t)ROWS * HIDD * sizeof(float));
    float* y2      = (float*)alloc((size_t)ROWS * HIDD * sizeof(float));
    float* ubuf    = (float*)alloc((size_t)ROWS * 2 * HIDD * sizeof(float));
    float* dtmp    = (float*)alloc((size_t)ROWS * HIDD * sizeof(float));

    prep_kernel<<<(ROWS * NBORD + 255) / 256, 256, 0, stream>>>(
        conv0_w, conv1_w, wf0, wf1, pooled, h0);

    conv0_kernel<<<NTILE_TOT / 8, 256, 0, stream>>>(x, wf0, conv0_b, ln_c0_s, ln_c0_b, h0);
    conv1_kernel<<<NTILE_TOT / 16, 256, 0, stream>>>(h0, wf1, conv1_b, ln_c1_s, ln_c1_b, pooled);

    // proj + LN  (mean pool: divide sums by 7056 via aScale)
    gemm_f32<<<(ROWS * HIDD + 255) / 256, 256, 0, stream>>>(
        pooled, proj_w, proj_b, nullptr, hpre, ROWS, HIDD, C1, 1.f / (float)PIX, 0);
    ln256_kernel<<<ROWS, 256, 0, stream>>>(hpre, ln_p_s, ln_p_b, nullptr, h);

    for (int i = 0; i < 2; ++i) {
        gemm_f32<<<(ROWS * 3 * HIDD + 255) / 256, 256, 0, stream>>>(
            h, qkv_w + (size_t)i * HIDD * 3 * HIDD, qkv_b + (size_t)i * 3 * HIDD,
            nullptr, qkv, ROWS, 3 * HIDD, HIDD, 1.f, 0);
        attn_kernel<<<BATCH * NHEAD * SEQ, 64, 0, stream>>>(qkv, pad, ybuf);
        gemm_f32<<<(ROWS * HIDD + 255) / 256, 256, 0, stream>>>(
            ybuf, out_w + (size_t)i * HIDD * HIDD, out_b + (size_t)i * HIDD,
            pad, y2, ROWS, HIDD, HIDD, 1.f, 0);
        ln256_kernel<<<ROWS, 256, 0, stream>>>(
            y2, ln_a_s + (size_t)i * HIDD, ln_a_b + (size_t)i * HIDD, h, h);
        gemm_f32<<<(ROWS * 2 * HIDD + 255) / 256, 256, 0, stream>>>(
            h, up_w + (size_t)i * HIDD * 2 * HIDD, nullptr, nullptr,
            ubuf, ROWS, 2 * HIDD, HIDD, 1.f, 1);
        gemm_f32<<<(ROWS * HIDD + 255) / 256, 256, 0, stream>>>(
            ubuf, down_w + (size_t)i * 2 * HIDD * HIDD, nullptr, nullptr,
            dtmp, ROWS, HIDD, 2 * HIDD, 1.f, 0);
        ln256_kernel<<<ROWS, 256, 0, stream>>>(
            dtmp, ln_m_s + (size_t)i * HIDD, ln_m_b + (size_t)i * HIDD, h, h);
    }

    gemm_f32<<<(ROWS * 18 + 255) / 256, 256, 0, stream>>>(
        h, head_w, head_b, nullptr, (float*)d_out, ROWS, 18, HIDD, 1.f, 0);
}
